// Attention_43963285242510
// MI455X (gfx1250) — matrix operand order repaired and checked by reading, not compiled, not run on hardware
//
#include <hip/hip_runtime.h>
#include <hip/hip_bf16.h>

typedef __attribute__((ext_vector_type(16))) __bf16 v16bf;
typedef __attribute__((ext_vector_type(8)))  __bf16 v8bf;
typedef __attribute__((ext_vector_type(4)))  __bf16 v4bf;
typedef __attribute__((ext_vector_type(8)))  float  v8f;

// exact spelling from the hipcc diagnostic for the async-LDS builtin params
typedef int v4i __attribute__((vector_size(4 * sizeof(int))));
typedef __attribute__((address_space(1))) v4i gv4i;   // printed as "__device__"
typedef __attribute__((address_space(3))) v4i lv4i;   // printed as "__shared__"

#define B_   2
#define L_   2048
#define D_   2048
#define H_   16
#define KV_  8
#define HD_  128
#define REP_ (H_ / KV_)
#define SCALE_ 0.08838834764831845f   /* 128^-0.5 */
#define EPS_   1e-5f
#define ROPE_BASE_ 10000.0f           /* alpha==1 -> base unchanged */

// ---------------------------------------------------------------------------
// CDNA5 async global->LDS copy path (ASYNCcnt). Guarded so the file always
// compiles; fallback is a plain b128 load + ds store.
// ---------------------------------------------------------------------------
#if defined(__HIP_DEVICE_COMPILE__) && __has_builtin(__builtin_amdgcn_global_load_async_to_lds_b128)
#define ASYNC_LDS 1
#else
#define ASYNC_LDS 0
#endif

static __device__ __forceinline__ void cp16_async(__bf16* lds, const __bf16* g) {
#if ASYNC_LDS
  __builtin_amdgcn_global_load_async_to_lds_b128(
      (gv4i*)(unsigned long long)(uintptr_t)g,     // global: flat addr == AS1 addr
      (lv4i*)(unsigned)(uintptr_t)lds,             // LDS: low 32 bits == LDS offset
      0, 0);
#else
  *(v8bf*)lds = *(const v8bf*)g;
#endif
}

static __device__ __forceinline__ void async_fence() {
#if ASYNC_LDS
#if __has_builtin(__builtin_amdgcn_s_wait_asynccnt)
  __builtin_amdgcn_s_wait_asynccnt(0);
#else
  asm volatile("s_wait_asynccnt 0x0" ::: "memory");
#endif
#endif
}

static __device__ __forceinline__ __bf16 f2bf(float f) {
  unsigned u = __builtin_bit_cast(unsigned, f);
  unsigned r = (u + 0x7FFFu + ((u >> 16) & 1u)) >> 16;   // round-to-nearest-even
  return __builtin_bit_cast(__bf16, (unsigned short)r);
}

static __device__ __forceinline__ void split2(unsigned u, __bf16& lo, __bf16& hi) {
  lo = __builtin_bit_cast(__bf16, (unsigned short)(u & 0xffffu));
  hi = __builtin_bit_cast(__bf16, (unsigned short)(u >> 16));
}

// Build a 16-element bf16 A/B fragment from two 16-byte LDS reads.
static __device__ __forceinline__ v16bf pack16(const __bf16* lo, const __bf16* hi) {
  v8bf a = *(const v8bf*)lo;
  v8bf b = *(const v8bf*)hi;
  v16bf r;
#pragma unroll
  for (int i = 0; i < 8; ++i) { r[i] = a[i]; r[i + 8] = b[i]; }
  return r;
}

// ---------------------------------------------------------------------------
// One-time fp32 -> bf16 conversion (vectorized x4). All sizes are %1024.
// ---------------------------------------------------------------------------
__global__ __launch_bounds__(256)
void cvt_f32_bf16(const float* __restrict__ in, __bf16* __restrict__ out, int n4) {
  int i = blockIdx.x * 256 + threadIdx.x;
  if (i >= n4) return;
  float4 f = ((const float4*)in)[i];
  v4bf o;
  o[0] = f2bf(f.x); o[1] = f2bf(f.y); o[2] = f2bf(f.z); o[3] = f2bf(f.w);
  ((v4bf*)out)[i] = o;
}

// ---------------------------------------------------------------------------
// Tiled GEMM: C[M,N] = A[M,K] @ B[K,N], bf16 in, fp32 or bf16 out.
// 128 threads (4 waves), block tile 64x64, K-step 32. A staged via async
// global->LDS copies; B staged transposed ([n][k]) via b32 loads.
// ---------------------------------------------------------------------------
template <bool OUT_BF16>
__global__ __launch_bounds__(128)
void gemm_bf16(const __bf16* __restrict__ A, const __bf16* __restrict__ Bw,
               void* __restrict__ Cout, int M, int N, int K) {
  __shared__ __bf16 sA[64][32];   // [m][k]
  __shared__ __bf16 sB[64][32];   // [n][k]

  const int tid  = threadIdx.x;
  const int wave = tid >> 5, lane = tid & 31;
  const int half = lane >> 4, l16 = lane & 15;
  const int m0 = blockIdx.y * 64;
  const int n0 = blockIdx.x * 64;

  v8f acc[4] = {};

  for (int k0 = 0; k0 < K; k0 += 32) {
    __syncthreads();
    // A tile: 64 rows x 32 bf16 = 256 x 16B segments, async copy
    for (int s = tid; s < 256; s += 128) {
      int r = s >> 2, c = (s & 3) * 8;
      cp16_async(&sA[r][c], &A[(size_t)(m0 + r) * K + k0 + c]);
    }
    // B tile transposed: each thread moves pairs (k, n..n+1)
    for (int i = tid; i < 1024; i += 128) {
      int kk = i >> 5, n2 = (i & 31) * 2;
      unsigned u = *(const unsigned*)&Bw[(size_t)(k0 + kk) * N + n0 + n2];
      split2(u, sB[n2][kk], sB[n2 + 1][kk]);
    }
    if (k0 + 32 < K) {                       // prefetch next tiles into L2
      __builtin_prefetch(&A[(size_t)(m0 + (tid & 63)) * K + k0 + 32], 0, 0);
      __builtin_prefetch(&Bw[(size_t)(k0 + 32 + (tid & 31)) * N + n0], 0, 0);
    }
    async_fence();
    __syncthreads();

    const int arow = 16 * wave + l16;
    // A frag: lanes0-15 K=0..15 ; lanes16-31 K=16..31 (contiguous)
    v16bf a = pack16(&sA[arow][half * 16], &sA[arow][half * 16 + 8]);
#pragma unroll
    for (int t = 0; t < 4; ++t) {
      const int n = 16 * t + l16;
      // B frag: lanes0-15 K=0..15 ; lanes16-31 K=16..31 (contiguous)
      v16bf b = pack16(&sB[n][half * 16], &sB[n][half * 16 + 8]);
      acc[t] = __builtin_amdgcn_wmma_f32_16x16x32_bf16(
          false, a, false, b, (short)0, acc[t], false, false);
    }
  }

#pragma unroll
  for (int t = 0; t < 4; ++t)
#pragma unroll
    for (int r = 0; r < 8; ++r) {
      int row = m0 + 16 * wave + (half ? r + 8 : r);    // C layout (ISA 7.12.2)
      int col = n0 + 16 * t + l16;
      if (OUT_BF16) ((__bf16*)Cout)[(size_t)row * N + col] = f2bf(acc[t][r]);
      else          ((float*)Cout)[(size_t)row * N + col] = acc[t][r];
    }
}

// ---------------------------------------------------------------------------
// RoPE + RMSNorm, one wave32 per (token,head) 128-vector. fp32 in, bf16 out.
// Pairs (i, i+64): lane owns elements lane, lane+32 (x1) / lane+64, lane+96 (x2).
// ---------------------------------------------------------------------------
__global__ __launch_bounds__(128)
void rope_rms(const float* __restrict__ in, __bf16* __restrict__ out,
              const float* __restrict__ w, int nheads) {
  const int wave = threadIdx.x >> 5, lane = threadIdx.x & 31;
  const int gid  = blockIdx.x * 4 + wave;
  const int head = gid % nheads;
  const int row  = gid / nheads;        // b*L + l
  const int pos  = row % L_;

  const float* v = in  + (size_t)row * (nheads * HD_) + head * HD_;
  __bf16*      o = out + (size_t)row * (nheads * HD_) + head * HD_;

  float x1a = v[lane],      x1b = v[lane + 32];
  float x2a = v[lane + 64], x2b = v[lane + 96];

  float fa = __powf(ROPE_BASE_, (float)lane        * (1.0f / 64.0f));
  float fb = __powf(ROPE_BASE_, (float)(lane + 32) * (1.0f / 64.0f));
  float sa, ca, sb, cb;
  __sincosf((float)pos * fa, &sa, &ca);
  __sincosf((float)pos * fb, &sb, &cb);

  float o0 = x1a * ca - x2a * sa;
  float o1 = x1b * cb - x2b * sb;
  float o2 = x2a * ca + x1a * sa;
  float o3 = x2b * cb + x1b * sb;

  float ss = o0 * o0 + o1 * o1 + o2 * o2 + o3 * o3;
#pragma unroll
  for (int m = 1; m < 32; m <<= 1) ss += __shfl_xor(ss, m, 32);
  float inv = rsqrtf(ss * (1.0f / HD_) + EPS_);

  o[lane]      = f2bf(o0 * inv * w[lane]);
  o[lane + 32] = f2bf(o1 * inv * w[lane + 32]);
  o[lane + 64] = f2bf(o2 * inv * w[lane + 64]);
  o[lane + 96] = f2bf(o3 * inv * w[lane + 96]);
}

// ---------------------------------------------------------------------------
// Flash attention (GQA), bf16 Q/K/V in, bf16 O out. Block = (b, h, 64 queries),
// wave = 16 queries, key blocks of 32. Q/K staged via async copies; V staged
// transposed. Scores scaled by HD^-1/2 inside the online softmax.
// ---------------------------------------------------------------------------
__global__ __launch_bounds__(128)
void flash_attn(const __bf16* __restrict__ q, const __bf16* __restrict__ kbuf,
                const __bf16* __restrict__ vbuf, __bf16* __restrict__ o) {
  __shared__ __bf16 sQ[64][HD_];        // [q][d]
  __shared__ __bf16 sK[32][HD_];        // [key][d]
  __shared__ __bf16 sVt[HD_][32];       // [d][key]
  __shared__ __bf16 sP[4][16][32];      // per-wave P tile (C-layout -> A-layout)

  const int tid  = threadIdx.x;
  const int wave = tid >> 5, lane = tid & 31;
  const int half = lane >> 4, l16 = lane & 15;
  const int q0  = blockIdx.x * 64;
  const int b   = blockIdx.y / H_;
  const int h   = blockIdx.y % H_;
  const int kvh = h / REP_;
  const int qld = H_ * HD_, kld = KV_ * HD_;

  // Q tile: 64 x 128 bf16 = 1024 x 16B segments, async
  for (int s = tid; s < 1024; s += 128) {
    int r = s >> 4, c = (s & 15) * 8;
    cp16_async(&sQ[r][c], &q[(size_t)(b * L_ + q0 + r) * qld + h * HD_ + c]);
  }

  v8f acc[8] = {};
  float mrow[8], lrow[8];
#pragma unroll
  for (int r = 0; r < 8; ++r) { mrow[r] = -1e30f; lrow[r] = 0.0f; }

  const int arow = 16 * wave + l16;

  for (int kb = 0; kb < L_; kb += 32) {
    __syncthreads();                                   // prev readers done
    // K tile: 32 x 128 = 512 x 16B segments, async
    for (int s = tid; s < 512; s += 128) {
      int r = s >> 4, c = (s & 15) * 8;
      cp16_async(&sK[r][c], &kbuf[(size_t)(b * L_ + kb + r) * kld + kvh * HD_ + c]);
    }
    // V tile transposed: pairs (k, d..d+1)
    for (int i = tid; i < 2048; i += 128) {
      int kk = i >> 6, d2 = (i & 63) * 2;
      unsigned u = *(const unsigned*)&vbuf[(size_t)(b * L_ + kb + kk) * kld + kvh * HD_ + d2];
      split2(u, sVt[d2][kk], sVt[d2 + 1][kk]);
    }
    async_fence();
    __syncthreads();

    // S = Q K^T : two 16x16 key tiles, K-dim = HD in 4 chunks of 32
    v8f s0 = {}, s1 = {};
#pragma unroll
    for (int d0 = 0; d0 < HD_; d0 += 32) {
      v16bf a  = pack16(&sQ[arow][d0 + half * 16], &sQ[arow][d0 + half * 16 + 8]);
      v16bf b0 = pack16(&sK[l16][d0 + half * 16],      &sK[l16][d0 + half * 16 + 8]);
      v16bf b1 = pack16(&sK[16 + l16][d0 + half * 16], &sK[16 + l16][d0 + half * 16 + 8]);
      s0 = __builtin_amdgcn_wmma_f32_16x16x32_bf16(false, a, false, b0, (short)0, s0, false, false);
      s1 = __builtin_amdgcn_wmma_f32_16x16x32_bf16(false, a, false, b1, (short)0, s1, false, false);
    }

    // online softmax; xor masks 1..8 stay within a 16-lane half (rows r / r+8)
#pragma unroll
    for (int r = 0; r < 8; ++r) {
      float v0 = s0[r] * SCALE_, v1 = s1[r] * SCALE_;
      float lm = fmaxf(v0, v1);
#pragma unroll
      for (int m = 1; m < 16; m <<= 1) lm = fmaxf(lm, __shfl_xor(lm, m, 32));
      float mn    = fmaxf(mrow[r], lm);
      float alpha = __expf(mrow[r] - mn);
      float p0 = __expf(v0 - mn);
      float p1 = __expf(v1 - mn);
      float rs = p0 + p1;
#pragma unroll
      for (int m = 1; m < 16; m <<= 1) rs += __shfl_xor(rs, m, 32);
      lrow[r] = lrow[r] * alpha + rs;
      mrow[r] = mn;
#pragma unroll
      for (int t = 0; t < 8; ++t) acc[t][r] *= alpha;
      int prw = half ? r + 8 : r;
      sP[wave][prw][l16]      = f2bf(p0);
      sP[wave][prw][l16 + 16] = f2bf(p1);
    }

    // intra-wave LDS RAW: lanes read P written by other lanes of same wave
    asm volatile("s_wait_dscnt 0" ::: "memory");

    // O += P V : A = P (16x32), B = V^T columns, 8 d-tiles of 16
    v16bf ap = pack16(&sP[wave][l16][half * 16], &sP[wave][l16][half * 16 + 8]);
#pragma unroll
    for (int t = 0; t < 8; ++t) {
      v16bf bv = pack16(&sVt[16 * t + l16][half * 16], &sVt[16 * t + l16][half * 16 + 8]);
      acc[t] = __builtin_amdgcn_wmma_f32_16x16x32_bf16(
          false, ap, false, bv, (short)0, acc[t], false, false);
    }
  }

#pragma unroll
  for (int r = 0; r < 8; ++r) {
    float inv = 1.0f / lrow[r];
    int row = q0 + 16 * wave + (half ? r + 8 : r);
#pragma unroll
    for (int t = 0; t < 8; ++t)
      o[(size_t)(b * L_ + row) * qld + h * HD_ + 16 * t + l16] = f2bf(acc[t][r] * inv);
  }
}

// ---------------------------------------------------------------------------
extern "C" void kernel_launch(void* const* d_in, const int* in_sizes, int n_in,
                              void* d_out, int out_size, void* d_ws, size_t ws_size,
                              hipStream_t stream) {
  (void)in_sizes; (void)n_in; (void)out_size; (void)ws_size;
  const float* x  = (const float*)d_in[0];
  const float* wq = (const float*)d_in[1];
  const float* wk = (const float*)d_in[2];
  const float* wv = (const float*)d_in[3];
  const float* wo = (const float*)d_in[4];
  const float* qn = (const float*)d_in[5];
  const float* kn = (const float*)d_in[6];
  float* out = (float*)d_out;

  const int M = B_ * L_;                               // 4096 tokens
  const size_t nX  = (size_t)M * D_;                   // 8.4M
  const size_t nWq = (size_t)D_ * (H_ * HD_);          // 4.2M
  const size_t nWk = (size_t)D_ * (KV_ * HD_);         // 2.1M
  const size_t nQ  = (size_t)M * (H_ * HD_);
  const size_t nK  = (size_t)M * (KV_ * HD_);

  __bf16* xb   = (__bf16*)d_ws;
  __bf16* wqb  = xb   + nX;
  __bf16* wkb  = wqb  + nWq;
  __bf16* wvb  = wkb  + nWk;
  __bf16* wob  = wvb  + nWk;
  __bf16* q16  = wob  + nWq;
  __bf16* k16  = q16  + nQ;
  __bf16* v16  = k16  + nK;
  __bf16* ao16 = v16  + nK;
  float*  qf   = (float*)(ao16 + nQ);
  float*  kf   = qf + nQ;

  dim3 blk(128);
  // one-time bf16 conversions (L2-resident afterwards: 192MB global L2)
  cvt_f32_bf16<<<(int)(nX  / 1024), 256, 0, stream>>>(x,  xb,  (int)(nX  / 4));
  cvt_f32_bf16<<<(int)(nWq / 1024), 256, 0, stream>>>(wq, wqb, (int)(nWq / 4));
  cvt_f32_bf16<<<(int)(nWk / 1024), 256, 0, stream>>>(wk, wkb, (int)(nWk / 4));
  cvt_f32_bf16<<<(int)(nWk / 1024), 256, 0, stream>>>(wv, wvb, (int)(nWk / 4));
  cvt_f32_bf16<<<(int)(nWq / 1024), 256, 0, stream>>>(wo, wob, (int)(nWq / 4));

  gemm_bf16<false><<<dim3((H_ * HD_) / 64,  M / 64), blk, 0, stream>>>(xb, wqb, qf,  M, H_ * HD_,  D_);
  gemm_bf16<false><<<dim3((KV_ * HD_) / 64, M / 64), blk, 0, stream>>>(xb, wkb, kf,  M, KV_ * HD_, D_);
  gemm_bf16<true ><<<dim3((KV_ * HD_) / 64, M / 64), blk, 0, stream>>>(xb, wvb, v16, M, KV_ * HD_, D_);

  rope_rms<<<(M * H_)  / 4, blk, 0, stream>>>(qf, q16, qn, H_);
  rope_rms<<<(M * KV_) / 4, blk, 0, stream>>>(kf, k16, kn, KV_);

  flash_attn<<<dim3(L_ / 64, B_ * H_), blk, 0, stream>>>(q16, k16, v16, ao16);

  gemm_bf16<false><<<dim3(D_ / 64, M / 64), blk, 0, stream>>>(ao16, wob, out, M, D_, H_ * HD_);
}
